// LightCrossAttention_18562848653364
// MI455X (gfx1250) — matrix operand order repaired and checked by reading, not compiled, not run on hardware
//
#include <hip/hip_runtime.h>
#include <hip/hip_bf16.h>

// ---------------------------------------------------------------------------
// LightCrossAttention for MI455X (gfx1250, wave32, WMMA f32_16x16x32_f16)
// B=2, Nt=Nf=4096, C=256, H=4, Dh=64
// v4: fix async builtin arg types (int4 vectors in AS(1)/AS(3)).
//     async global->LDS K/V staging (ASYNCcnt path, 1 barrier/chunk),
//     softmax scale folded into Q fragments.
// ---------------------------------------------------------------------------

typedef _Float16 f16;
typedef __attribute__((ext_vector_type(16))) _Float16 v16h;
typedef __attribute__((ext_vector_type(8)))  float    v8f;
typedef __attribute__((ext_vector_type(4)))  float    v4f;
typedef int v4i __attribute__((vector_size(16)));

union AFu { v16h h; v4f f4[2]; };

#if defined(__AMDGCN__) && __has_builtin(__builtin_amdgcn_global_load_async_to_lds_b128)
#define ASYNC_LDS 1
#else
#define ASYNC_LDS 0
#endif

#if ASYNC_LDS
__device__ __forceinline__ void async_copy16(const f16* g, f16* l) {
  // GLOBAL_LOAD_ASYNC_TO_LDS_B128: per-lane 16B memory->LDS, tracked by ASYNCcnt
  __builtin_amdgcn_global_load_async_to_lds_b128(
      (__attribute__((address_space(1))) v4i*)(g),
      (__attribute__((address_space(3))) v4i*)(l), 0, 0);
}
__device__ __forceinline__ void wait_async0() {
#if __has_builtin(__builtin_amdgcn_s_wait_asynccnt)
  __builtin_amdgcn_s_wait_asynccnt(0);
#else
  asm volatile("s_wait_asynccnt 0" ::: "memory");
#endif
}
#endif

__device__ __forceinline__ v8f zero8() {
  v8f z;
#pragma unroll
  for (int e = 0; e < 8; ++e) z[e] = 0.0f;
  return z;
}

__device__ __forceinline__ v8f wmma_f16(v16h a, v16h b, v8f c) {
  // v_wmma_f32_16x16x32_f16 : D = A(16x32 f16) * B(32x16 f16) + C(16x16 f32)
  return __builtin_amdgcn_wmma_f32_16x16x32_f16(
      /*neg_a=*/false, a, /*neg_b=*/false, b,
      /*c_mod=*/(short)0, c, /*reuse_a=*/false, /*reuse_b=*/false);
}

// A-fragment (16x32 f16, row-major source, leading dim ld in elems).
// ISA layout: lane l -> row m=l&15, kb=(l>>4)*8; VGPR0-3 = K[kb..kb+7],
// VGPR4-7 = K[kb+16..kb+23]  (two contiguous 16-byte loads).
__device__ __forceinline__ v16h load_frag_a(const f16* base, int ld, int lane) {
  const int m  = lane & 15;
  const int kb = (lane >> 4) * 8;
  const f16* p = base + (size_t)m * ld + kb;
  AFu u;
  u.f4[0] = *(const v4f*)(p);
  u.f4[1] = *(const v4f*)(p + 16);
  return u.h;
}

// B-fragment (32x16 f16). Source layout [n][k] ("transposed"), ld = k-stride.
// ISA layout: lane l -> col n=l&15, kb=(l>>4)*8; K[kb..kb+7], K[kb+16..kb+23].
__device__ __forceinline__ v16h load_frag_b(const f16* base, int ld, int lane) {
  const int n  = lane & 15;
  const int kb = (lane >> 4) * 8;
  const f16* p = base + (size_t)n * ld + kb;
  AFu u;
  u.f4[0] = *(const v4f*)(p);
  u.f4[1] = *(const v4f*)(p + 16);
  return u.h;
}

// 16x64 output strip: A row-major [m][k], B transposed [n][k], K % 32 == 0.
// One A-fragment feeds 4 B-fragments / 4 WMMAs per k-step (A reuse 4x).
__device__ __forceinline__ void gemm_tile4(const f16* A, int lda,
                                           const f16* B, int ldb,
                                           int K, int lane, v8f acc[4]) {
  for (int k = 0; k < K; k += 32) {
    v16h a = load_frag_a(A + k, lda, lane);
#pragma unroll
    for (int j = 0; j < 4; ++j) {
      v16h bj = load_frag_b(B + (size_t)(j * 16) * ldb + k, ldb, lane);
      acc[j] = wmma_f16(a, bj, acc[j]);
    }
  }
}

// 16-lane (row-group) reductions for C-layout rows (wave32; masks < 16 stay in half)
__device__ __forceinline__ void rowmax16(v8f& t) {
#pragma unroll
  for (int off = 1; off < 16; off <<= 1) {
#pragma unroll
    for (int e = 0; e < 8; ++e) t[e] = fmaxf(t[e], __shfl_xor(t[e], off, 32));
  }
}
__device__ __forceinline__ void rowsum16(v8f& t) {
#pragma unroll
  for (int off = 1; off < 16; off <<= 1) {
#pragma unroll
    for (int e = 0; e < 8; ++e) t[e] += __shfl_xor(t[e], off, 32);
  }
}

// ---------------------------------------------------------------------------
// 1) RMS norm:  y = x * rsqrt(mean(x^2)+eps) * w ; writes f32 (opt) and f16
// ---------------------------------------------------------------------------
__global__ __launch_bounds__(256) void rmsnorm_kernel(
    const float* __restrict__ x, const float* __restrict__ w,
    float* __restrict__ y32, f16* __restrict__ y16) {
  const int row = blockIdx.x;
  const int c   = threadIdx.x;            // C = 256
  const float v = x[(size_t)row * 256 + c];
  __shared__ float red[256];
  red[c] = v * v;
  __syncthreads();
#pragma unroll
  for (int s = 128; s > 0; s >>= 1) {
    if (c < s) red[c] += red[c + s];
    __syncthreads();
  }
  const float norm = rsqrtf(red[0] * (1.0f / 256.0f) + 1e-5f);
  const float o = v * norm * w[c];
  if (y32) y32[(size_t)row * 256 + c] = o;
  y16[(size_t)row * 256 + c] = (f16)o;
}

// ---------------------------------------------------------------------------
// 2) Weight convert + transpose: in f32 [K][N] -> out f16 [N][K]
// ---------------------------------------------------------------------------
__global__ __launch_bounds__(256) void transpose_w_kernel(
    const float* __restrict__ in, f16* __restrict__ out, int K, int N) {
  const int idx = blockIdx.x * 256 + threadIdx.x;
  if (idx >= K * N) return;
  const int k = idx / N, n = idx % N;
  out[(size_t)n * K + k] = (f16)in[idx];
}

// ---------------------------------------------------------------------------
// 3) Q projection: q = tn @ wq + bq  (M=8192, N=256, K=256) -> f16
//    One wave = 16x64 strip. 2048 waves -> 256 blocks.
// ---------------------------------------------------------------------------
__global__ __launch_bounds__(256) void gemm_q_kernel(
    const f16* __restrict__ A, const f16* __restrict__ Bw,
    const float* __restrict__ bias, f16* __restrict__ out) {
  const int lane = threadIdx.x & 31, widx = threadIdx.x >> 5;
  const int w = blockIdx.x * 8 + widx;
  const int mt = w >> 2, nt = w & 3;        // N/64 = 4
  const int Mb = mt << 4, Nb = nt << 6;
  v8f acc[4];
#pragma unroll
  for (int j = 0; j < 4; ++j) acc[j] = zero8();
  gemm_tile4(A + (size_t)Mb * 256, 256, Bw + (size_t)Nb * 256, 256, 256, lane, acc);
  const int half = lane >> 4;
#pragma unroll
  for (int j = 0; j < 4; ++j) {
    const int n = Nb + j * 16 + (lane & 15);
    const float bv = bias[n];
#pragma unroll
    for (int r = 0; r < 8; ++r) {
      const int row = Mb + r + 8 * half;
      out[(size_t)row * 256 + n] = (f16)(acc[j][r] + bv);
    }
  }
}

// ---------------------------------------------------------------------------
// 4) KV projection: kv = fn @ wkv + bkv  (M=8192, N=512, K=256)
//    n<256 -> K row-major [row][chan] ; n>=256 -> V transposed [b][h][d][f]
//    One wave = 16x64 strip. 4096 waves -> 512 blocks.
// ---------------------------------------------------------------------------
__global__ __launch_bounds__(256) void gemm_kv_kernel(
    const f16* __restrict__ A, const f16* __restrict__ Bw,
    const float* __restrict__ bias, f16* __restrict__ kh, f16* __restrict__ vt) {
  const int lane = threadIdx.x & 31, widx = threadIdx.x >> 5;
  const int w = blockIdx.x * 8 + widx;
  const int mt = w >> 3, nt = w & 7;        // N/64 = 8
  const int Mb = mt << 4, Nb = nt << 6;
  v8f acc[4];
#pragma unroll
  for (int j = 0; j < 4; ++j) acc[j] = zero8();
  gemm_tile4(A + (size_t)Mb * 256, 256, Bw + (size_t)Nb * 256, 256, 256, lane, acc);
  const int half = lane >> 4;
#pragma unroll
  for (int j = 0; j < 4; ++j) {
    const int n = Nb + j * 16 + (lane & 15);
    const float bv = bias[n];
#pragma unroll
    for (int r = 0; r < 8; ++r) {
      const int row = Mb + r + 8 * half;     // row = b*4096 + f
      const float val = acc[j][r] + bv;
      if (n < 256) {
        kh[(size_t)row * 256 + n] = (f16)val;
      } else {
        const int chan = n - 256;
        const int h = chan >> 6, d = chan & 63;
        const int bb = row >> 12, ff = row & 4095;
        vt[((size_t)((bb * 4 + h) * 64 + d)) * 4096 + ff] = (f16)val;
      }
    }
  }
}

// ---------------------------------------------------------------------------
// 5) Flash attention. One block = one (b,h) x 128 Q rows (8 waves x 16 rows).
//    K/V chunks (32 f-positions, 8 KB) staged into double-buffered LDS with
//    GLOBAL_LOAD_ASYNC_TO_LDS_B128 (ASYNCcnt); one barrier per chunk, next
//    chunk's async loads overlap the current chunk's compute.
//    Per chunk per wave: S (4 WMMA) -> online softmax -> P via LDS -> O (4 WMMA).
// ---------------------------------------------------------------------------
__global__ __launch_bounds__(256) void flash_attn_kernel(
    const f16* __restrict__ qh, const f16* __restrict__ kh,
    const f16* __restrict__ vt, f16* __restrict__ ao) {
  const int tid  = threadIdx.x;
  const int lane = tid & 31, widx = tid >> 5;
  const int blk  = blockIdx.x;               // 256 blocks
  const int b    = blk >> 7;                 // 128 blocks per batch
  const int rem  = blk & 127;
  const int h    = rem >> 5;                 // 32 blocks per head
  const int tblk = (rem & 31) << 7;          // 128 rows per block
  const int tbase = tblk + widx * 16;

  __shared__ f16 Kbuf[2][32 * 64];           // [buf][f_local][chan]  8 KB
  __shared__ f16 Vbuf[2][64 * 32];           // [buf][chan][f_local]  8 KB
  __shared__ f16 Plds[8][16 * 32];           // per-wave P            8 KB
  f16* pbuf = &Plds[widx][0];

  // cooperative staging addresses (each thread moves 16B of K and 16B of V)
  const int krow = tid >> 3, kcol = (tid & 7) * 8;   // 32 rows x 64 chan
  const int vrow = tid >> 2, vcol = (tid & 3) * 8;   // 64 chan x 32 f
  const f16* ksrc = kh + ((size_t)(b * 4096 + krow)) * 256 + h * 64 + kcol;
  const f16* vsrc = vt + ((size_t)((b * 4 + h) * 64 + vrow)) * 4096 + vcol;
  const int koff = krow * 64 + kcol;
  const int voff = vrow * 32 + vcol;

  // Q fragments with softmax scale (Dh^-0.5 = 0.125) folded in
  const f16* qbase = qh + ((size_t)(b * 4096 + tbase)) * 256 + h * 64;
  v16h qf0 = load_frag_a(qbase,      256, lane);   // channels 0..31
  v16h qf1 = load_frag_a(qbase + 32, 256, lane);   // channels 32..63
#pragma unroll
  for (int e = 0; e < 16; ++e) {
    qf0[e] = qf0[e] * (f16)0.125f;
    qf1[e] = qf1[e] * (f16)0.125f;
  }

  v8f O[4];
#pragma unroll
  for (int nb = 0; nb < 4; ++nb) O[nb] = zero8();
  v8f m, lsum;
#pragma unroll
  for (int e = 0; e < 8; ++e) { m[e] = -3.0e38f; lsum[e] = 0.0f; }

  const int colp = lane & 15;
  const int half = lane >> 4;

#if ASYNC_LDS
  // issue chunk 0 into buffer 0 (memory -> LDS, no VGPR roundtrip)
  async_copy16(ksrc, &Kbuf[0][koff]);
  async_copy16(vsrc, &Vbuf[0][voff]);
#else
  v4f kreg = *(const v4f*)(ksrc);
  v4f vreg = *(const v4f*)(vsrc);
#endif

  for (int fb = 0; fb < 4096; fb += 32) {
    const int cur = (fb >> 5) & 1;
#if ASYNC_LDS
    wait_async0();                           // my async portion landed in LDS
    __syncthreads();                         // everyone's portion landed;
                                             // prior compute on buf cur^1 done
    if (fb + 32 < 4096) {                    // overlap next chunk with compute
      async_copy16(ksrc + (size_t)(fb + 32) * 256, &Kbuf[cur ^ 1][koff]);
      async_copy16(vsrc + (fb + 32),               &Vbuf[cur ^ 1][voff]);
    }
#else
    __syncthreads();                         // previous chunk fully consumed
    *(v4f*)&Kbuf[cur][koff] = kreg;
    *(v4f*)&Vbuf[cur][voff] = vreg;
    __syncthreads();                         // staging visible to all waves
    if (fb + 32 < 4096) {
      kreg = *(const v4f*)(ksrc + (size_t)(fb + 32) * 256);
      vreg = *(const v4f*)(vsrc + (fb + 32));
    }
#endif
    const f16* kb = &Kbuf[cur][0];
    const f16* vb = &Vbuf[cur][0];

    // ---- S = Q K^T : two 16x16 tiles (f local 0..15 / 16..31), from LDS ----
    v16h k0a = load_frag_b(kb,             64, lane);  // chan 0..31
    v16h k0b = load_frag_b(kb + 32,        64, lane);  // chan 32..63
    v16h k1a = load_frag_b(kb + 1024,      64, lane);
    v16h k1b = load_frag_b(kb + 1024 + 32, 64, lane);
    v8f S0 = wmma_f16(qf1, k0b, wmma_f16(qf0, k0a, zero8()));
    v8f S1 = wmma_f16(qf1, k1b, wmma_f16(qf0, k1a, zero8()));

    // ---- online softmax (rows live across 16 lanes of the C layout) ----
    v8f cm;
#pragma unroll
    for (int e = 0; e < 8; ++e) cm[e] = fmaxf(S0[e], S1[e]);
    rowmax16(cm);
    v8f al;
#pragma unroll
    for (int e = 0; e < 8; ++e) {
      const float mn = fmaxf(m[e], cm[e]);
      al[e] = __expf(m[e] - mn);
      m[e] = mn;
    }
#pragma unroll
    for (int e = 0; e < 8; ++e) {
      lsum[e] *= al[e];
      O[0][e] *= al[e]; O[1][e] *= al[e]; O[2][e] *= al[e]; O[3][e] *= al[e];
    }
    v8f P0, P1, rs;
#pragma unroll
    for (int e = 0; e < 8; ++e) {
      P0[e] = __expf(S0[e] - m[e]);
      P1[e] = __expf(S1[e] - m[e]);
      rs[e] = P0[e] + P1[e];
    }
    rowsum16(rs);
#pragma unroll
    for (int e = 0; e < 8; ++e) lsum[e] += rs[e];

    // ---- C layout -> A-fragment layout via per-wave LDS (16x32 f16) ----
#pragma unroll
    for (int r = 0; r < 8; ++r) {
      const int row = r + 8 * half;
      pbuf[row * 32 + colp]      = (f16)P0[r];
      pbuf[row * 32 + colp + 16] = (f16)P1[r];
    }
    const v16h pf = load_frag_a(pbuf, 32, lane);

    // ---- O += P @ V : 4 WMMAs over the 64 head channels (V from LDS) ----
#pragma unroll
    for (int nb = 0; nb < 4; ++nb) {
      v16h vf = load_frag_b(vb + nb * 16 * 32, 32, lane);
      O[nb] = wmma_f16(pf, vf, O[nb]);
    }
  }

  // ---- normalize and store f16 attention output [row][h*64+chan] ----
  v8f inv;
#pragma unroll
  for (int e = 0; e < 8; ++e) inv[e] = 1.0f / lsum[e];
#pragma unroll
  for (int nb = 0; nb < 4; ++nb) {
    const int col = h * 64 + nb * 16 + colp;
#pragma unroll
    for (int r = 0; r < 8; ++r) {
      const int row = b * 4096 + tbase + r + 8 * half;
      ao[(size_t)row * 256 + col] = (f16)(O[nb][r] * inv[r]);
    }
  }
}

// ---------------------------------------------------------------------------
// 6) Output projection: out = ao @ wp + bp + tn   (f32 result)
// ---------------------------------------------------------------------------
__global__ __launch_bounds__(256) void gemm_p_kernel(
    const f16* __restrict__ A, const f16* __restrict__ Bw,
    const float* __restrict__ bias, const float* __restrict__ resid,
    float* __restrict__ out) {
  const int lane = threadIdx.x & 31, widx = threadIdx.x >> 5;
  const int w = blockIdx.x * 8 + widx;
  const int mt = w >> 2, nt = w & 3;
  const int Mb = mt << 4, Nb = nt << 6;
  v8f acc[4];
#pragma unroll
  for (int j = 0; j < 4; ++j) acc[j] = zero8();
  gemm_tile4(A + (size_t)Mb * 256, 256, Bw + (size_t)Nb * 256, 256, 256, lane, acc);
  const int half = lane >> 4;
#pragma unroll
  for (int j = 0; j < 4; ++j) {
    const int n = Nb + j * 16 + (lane & 15);
    const float bv = bias[n];
#pragma unroll
    for (int r = 0; r < 8; ++r) {
      const int row = Mb + r + 8 * half;
      const size_t idx = (size_t)row * 256 + n;
      out[idx] = acc[j][r] + bv + resid[idx];
    }
  }
}

// ---------------------------------------------------------------------------
// Host launcher
// ---------------------------------------------------------------------------
extern "C" void kernel_launch(void* const* d_in, const int* in_sizes, int n_in,
                              void* d_out, int out_size, void* d_ws, size_t ws_size,
                              hipStream_t stream) {
  (void)in_sizes; (void)n_in; (void)out_size; (void)ws_size;
  const float* t   = (const float*)d_in[0];
  const float* f   = (const float*)d_in[1];
  const float* nw  = (const float*)d_in[2];
  const float* wq  = (const float*)d_in[3];
  const float* bq  = (const float*)d_in[4];
  const float* wkv = (const float*)d_in[5];
  const float* bkv = (const float*)d_in[6];
  const float* wp  = (const float*)d_in[7];
  const float* bp  = (const float*)d_in[8];
  float* out = (float*)d_out;

  // Workspace layout (needs ~34.1 MB)
  char* ws = (char*)d_ws;
  float* tn32 = (float*)(ws);                        //  8 MB : tn f32 (residual)
  f16* tnh  = (f16*)(ws + (size_t)(8  << 20));       //  4 MB : tn f16
  f16* fnh  = (f16*)(ws + (size_t)(12 << 20));       //  4 MB : fn f16
  f16* qh   = (f16*)(ws + (size_t)(16 << 20));       //  4 MB : q  f16 [row][c]
  f16* kh   = (f16*)(ws + (size_t)(20 << 20));       //  4 MB : k  f16 [row][c]
  f16* vt   = (f16*)(ws + (size_t)(24 << 20));       //  4 MB : v  f16 [b][h][d][f]
  f16* aoh  = (f16*)(ws + (size_t)(28 << 20));       //  4 MB : attn out f16
  f16* wqt  = (f16*)(ws + (size_t)(32 << 20));                      // 128 KB
  f16* wkvt = (f16*)(ws + (size_t)(32 << 20) + 256 * 256 * 2);      // 256 KB
  f16* wpt  = (f16*)(ws + (size_t)(32 << 20) + 256 * 256 * 2 + 256 * 512 * 2);

  // Weights: f32 [K][N] -> f16 [N][K] (B-fragment friendly)
  transpose_w_kernel<<<256, 256, 0, stream>>>(wq,  wqt,  256, 256);
  transpose_w_kernel<<<512, 256, 0, stream>>>(wkv, wkvt, 256, 512);
  transpose_w_kernel<<<256, 256, 0, stream>>>(wp,  wpt,  256, 256);

  // RMS norms (8192 rows each)
  rmsnorm_kernel<<<8192, 256, 0, stream>>>(t, nw, tn32, tnh);
  rmsnorm_kernel<<<8192, 256, 0, stream>>>(f, nw, nullptr, fnh);

  // Projections (16x64 strips per wave)
  gemm_q_kernel<<<256, 256, 0, stream>>>(tnh, wqt, bq, qh);           // 2048 waves
  gemm_kv_kernel<<<512, 256, 0, stream>>>(fnh, wkvt, bkv, kh, vt);    // 4096 waves

  // Flash attention: 256 blocks, one (b,h,128-row) slab each, async K/V -> LDS
  flash_attn_kernel<<<256, 256, 0, stream>>>(qh, kh, vt, aoh);

  // Output projection + bias + residual
  gemm_p_kernel<<<256, 256, 0, stream>>>(aoh, wpt, bp, tn32, out);
}
